// PointWiseMLP_23527830847989
// MI455X (gfx1250) — matrix operand order repaired and checked by reading, not compiled, not run on hardware
//
#include <hip/hip_runtime.h>
#include <math.h>

#define RADIUS   0.1f
#define R2       (RADIUS * RADIUS)
#define NS       32
#define N1C      8192
#define N2C      8192
#define CIN      64
#define MF       32
#define COUT     128
#define CIN0     131          // 3 + 2*CIN
#define K0TILES  5            // padded K = 160
#define A0P      168          // LDS pitch (halves), 16B aligned, conflict-padded
#define WP       40           // LDS pitch (halves) for 32-wide K
#define BQ       4            // queries per MLP block
#define EPSC     1e-5f
#define NTILES   (N2C / 256)

typedef __attribute__((ext_vector_type(16))) _Float16 v16h;
typedef __attribute__((ext_vector_type(8)))  _Float16 v8h;
typedef __attribute__((ext_vector_type(8)))  float    v8f;

// Low 32 bits of a generic shared-aperture address == LDS byte offset
// (see ISA 10.2; the aperture tag lives in the upper 32 address bits).
__device__ __forceinline__ unsigned lds_off(const void* p)
{
    return (unsigned)(unsigned long long)(size_t)p;
}

// ---------------------------------------------------------------------------
// Kernel 1: ball query with async double-buffered support tiles.
// One thread per query point.  Tile t+1 streams into LDS via
// global_load_async_to_lds_b128 (ASYNCcnt) while tile t is scanned.
// Scanning in index order == reference's sort-ascending + take-first-32.
// ---------------------------------------------------------------------------
__global__ __launch_bounds__(256) void bq_kernel(
    const float* __restrict__ qxyz, const float* __restrict__ sxyz,
    const int*   __restrict__ qmask, const int* __restrict__ smask,
    int* __restrict__ idxOut, int* __restrict__ cntOut)
{
    __shared__ __align__(16) float sxyzT[2][768];   // xyz interleaved, 3072B/buf
    __shared__ __align__(16) int   smT[2][256];     // masks, 1024B/buf

    const int tid = threadIdx.x;
    const int q   = blockIdx.x * 256 + tid;          // global query id
    const int b   = q / N1C;                         // uniform per block
    const float* sb  = sxyz  + (size_t)b * N2C * 3;
    const int*   smk = smask + (size_t)b * N2C;

    const float qx = qxyz[(size_t)q * 3 + 0];
    const float qy = qxyz[(size_t)q * 3 + 1];
    const float qz = qxyz[(size_t)q * 3 + 2];
    const int   qm = qmask[q];

    // Each thread owns one 16B chunk per tile: threads 0..191 -> xyz (192*16B),
    // threads 192..255 -> mask (64*16B).  All addresses 16B aligned.
    const bool     isx  = tid < 192;
    const unsigned coff = (unsigned)(isx ? tid : (tid - 192)) * 16u;

    auto issueCopy = [&](int bf, int j0) {
        const char* gsrc = isx ? (const char*)(sb + (size_t)j0 * 3)
                               : (const char*)(smk + j0);
        unsigned long long gaddr = (unsigned long long)(size_t)gsrc + coff;
        unsigned           laddr = (isx ? lds_off(&sxyzT[bf][0])
                                        : lds_off(&smT[bf][0])) + coff;
        asm volatile("global_load_async_to_lds_b128 %0, %1, off"
                     :: "v"(laddr), "v"(gaddr) : "memory");
    };

    int cnt = 0, first = 0;
    int* myIdx = idxOut + (size_t)q * NS;

    issueCopy(0, 0);                                  // prefetch tile 0
    for (int t = 0; t < NTILES; ++t) {
        const int bf = t & 1;
        issueCopy(bf ^ 1, ((t + 1) % NTILES) * 256);  // prefetch next tile
        // 2 async ops outstanding; waiting <=1 retires the older (in-order).
        asm volatile("s_wait_asynccnt 1" ::: "memory");
        __syncthreads();                              // publish all waves' chunks
        if (qm && cnt < NS) {
            const float* xp = sxyzT[bf];
            const int*   mp = smT[bf];
            const int j0 = t * 256;
            for (int jj = 0; jj < 256; ++jj) {
                const float dx = qx - xp[jj * 3 + 0];
                const float dy = qy - xp[jj * 3 + 1];
                const float dz = qz - xp[jj * 3 + 2];
                const float d2 = dx * dx + dy * dy + dz * dz;
                if (mp[jj] && d2 < R2) {
                    if (cnt == 0) first = j0 + jj;
                    myIdx[cnt] = j0 + jj;
                    if (++cnt == NS) break;
                }
            }
        }
        __syncthreads();                              // readers done before overwrite
    }
    for (int k = cnt; k < NS; ++k) myIdx[k] = (cnt > 0) ? first : 0;
    cntOut[q] = cnt;
}

// ---------------------------------------------------------------------------
// WMMA operand loaders (f16, 16x16x32).
// A (16xK): lanes 0-15 row M=lane, K=0..15 contiguous; lanes
//           16-31 K=16..31.                   -> two 16B LDS chunks per lane.
// B (Kx16, stored [n][k]): lane&15 = col; lanes<16 K=0..15 contiguous,
//           lanes>=16 K=16..31.               -> two 16B LDS chunks per lane.
// ---------------------------------------------------------------------------
__device__ __forceinline__ v16h ldsA(const _Float16* base, int pitch,
                                     int mbase, int ktile, int lane)
{
    const int row = mbase + (lane & 15);
    const int kb  = ktile * 32 + ((lane >> 4) << 4);
    const _Float16* p = base + row * pitch + kb;
    v16h a;
    ((v8h*)&a)[0] = *(const v8h*)(p);
    ((v8h*)&a)[1] = *(const v8h*)(p + 8);
    return a;
}
__device__ __forceinline__ v16h ldsB(const _Float16* base, int pitch,
                                     int nbase, int ktile, int lane)
{
    const int col = nbase + (lane & 15);
    const int kb  = ktile * 32 + ((lane >> 4) << 4);
    const _Float16* p = base + col * pitch + kb;
    v16h bm;
    ((v8h*)&bm)[0] = *(const v8h*)(p);
    ((v8h*)&bm)[1] = *(const v8h*)(p + 8);
    return bm;
}
__device__ __forceinline__ v8f wmma_f16(v16h a, v16h b, v8f c)
{
    return __builtin_amdgcn_wmma_f32_16x16x32_f16(false, a, false, b,
                                                  (short)0, c, false, false);
}

// ---------------------------------------------------------------------------
// Kernel 2: fused gather -> (131->32->32->128) WMMA MLP -> masked mean pool.
// Block = 128 threads (4 waves); wave w owns query w (rows 32w..32w+31).
// Layer-0 K dimension is PERMUTED so gather writes are all 16B v8h stores:
//   k 0..63  = center feature        (ref channel 3+k)
//   k 64..127= gathered - center     (ref channel 67+(k-64))
//   k 128..130 = rel_pos             (ref channel k-128)
//   k 131..167 = zero pad
// sW0 is staged with the same permutation, so the GEMM is unchanged.
// ---------------------------------------------------------------------------
__global__ __launch_bounds__(128) void mlp_kernel(
    const float* __restrict__ qxyz, const float* __restrict__ sxyz,
    const int*   __restrict__ qmask, const float* __restrict__ feat,
    const float* __restrict__ W0, const float* __restrict__ g0,
    const float* __restrict__ b0, const float* __restrict__ rm0,
    const float* __restrict__ rv0,
    const float* __restrict__ W1, const float* __restrict__ g1,
    const float* __restrict__ b1, const float* __restrict__ rm1,
    const float* __restrict__ rv1,
    const float* __restrict__ W2, const float* __restrict__ g2,
    const float* __restrict__ b2, const float* __restrict__ rm2,
    const float* __restrict__ rv2,
    const int* __restrict__ idxIn, const int* __restrict__ cntIn,
    float* __restrict__ out)
{
    __shared__ __align__(16) _Float16 sA0[128 * A0P];
    __shared__ __align__(16) _Float16 sA1[128 * WP];
    __shared__ __align__(16) _Float16 sA2[128 * WP];
    __shared__ __align__(16) _Float16 sW0[MF   * A0P];
    __shared__ __align__(16) _Float16 sW1[MF   * WP];
    __shared__ __align__(16) _Float16 sW2[COUT * WP];
    __shared__ float sAlpha[192], sBeta[192];

    const int tid  = threadIdx.x;
    const int lane = tid & 31;
    const int w    = tid >> 5;                 // wave id == local query id
    const int qg   = blockIdx.x * BQ + w;      // global query
    const int b    = qg / N1C;
    const int n1   = qg % N1C;

    // ---- folded BN coefficients --------------------------------------------
    for (int t = tid; t < 192; t += 128) {
        const float *gp, *bp, *rmp, *rvp; int c;
        if (t < 32)      { gp = g0; bp = b0; rmp = rm0; rvp = rv0; c = t; }
        else if (t < 64) { gp = g1; bp = b1; rmp = rm1; rvp = rv1; c = t - 32; }
        else             { gp = g2; bp = b2; rmp = rm2; rvp = rv2; c = t - 64; }
        const float al = gp[c] * (1.0f / sqrtf(rvp[c] + EPSC));
        sAlpha[t] = al;
        sBeta[t]  = bp[c] - rmp[c] * al;
    }
    // ---- weights into LDS as [out_ch][k] f16 (layer0 K-permuted) -----------
    for (int t = tid; t < MF * A0P; t += 128) {
        const int n = t / A0P, k = t % A0P;
        float wv = 0.0f;
        if (k < 64)       wv = W0[n * CIN0 + 3 + k];            // center block
        else if (k < 128) wv = W0[n * CIN0 + 67 + (k - 64)];    // diff block
        else if (k < CIN0) wv = W0[n * CIN0 + (k - 128)];       // rel_pos block
        sW0[t] = (_Float16)wv;
    }
    for (int t = tid; t < MF * WP; t += 128) {
        const int n = t / WP, k = t % WP;
        sW1[t] = (k < MF) ? (_Float16)W1[n * MF + k] : (_Float16)0.0f;
    }
    for (int t = tid; t < COUT * WP; t += 128) {
        const int n = t / WP, k = t % WP;
        sW2[t] = (k < MF) ? (_Float16)W2[n * MF + k] : (_Float16)0.0f;
    }

    // ---- gather: build permuted A0 row with 16B vector stores ---------------
    const int* myIdx = idxIn + (size_t)qg * NS;
    const int  is  = myIdx[lane];              // this sample's support index
    const int  i0  = myIdx[0];                 // center index (wave-uniform)
    const int  row = w * 32 + lane;
    _Float16* arow = sA0 + row * A0P;

    {
        const float qxv = qxyz[(size_t)qg * 3 + 0];
        const float qyv = qxyz[(size_t)qg * 3 + 1];
        const float qzv = qxyz[(size_t)qg * 3 + 2];
        const float* sp = sxyz + ((size_t)b * N2C + is) * 3;
        v8h relv = {};
        relv[0] = (_Float16)((sp[0] - qxv) * (1.0f / RADIUS));
        relv[1] = (_Float16)((sp[1] - qyv) * (1.0f / RADIUS));
        relv[2] = (_Float16)((sp[2] - qzv) * (1.0f / RADIUS));
        *(v8h*)(arow + 128) = relv;            // rel_pos + start of zero pad
        const v8h z = {};
        *(v8h*)(arow + 136) = z;
        *(v8h*)(arow + 144) = z;
        *(v8h*)(arow + 152) = z;
        *(v8h*)(arow + 160) = z;               // pads through col 167
    }
    for (int c0 = 0; c0 < CIN; c0 += 8) {
        v8h cvec, dvec;
#pragma unroll
        for (int j = 0; j < 8; ++j) {
            const float* fr = feat + ((size_t)b * CIN + (c0 + j)) * N2C;
            const float cf = fr[i0];           // broadcast (uniform addr)
            const float gv = fr[is];           // gathered
            cvec[j] = (_Float16)cf;
            dvec[j] = (_Float16)(gv - cf);
        }
        *(v8h*)(arow + c0)      = cvec;        // center block
        *(v8h*)(arow + 64 + c0) = dvec;        // diff block
    }
    __syncthreads();

    // ---- layer 0: 131(pad 160) -> 32 ---------------------------------------
    for (int mi = 0; mi < 2; ++mi) {
        const int mt = 2 * w + mi;
        for (int nt = 0; nt < 2; ++nt) {
            v8f acc = {};
            for (int kt = 0; kt < K0TILES; ++kt) {
                const v16h a  = ldsA(sA0, A0P, mt * 16, kt, lane);
                const v16h bm = ldsB(sW0, A0P, nt * 16, kt, lane);
                acc = wmma_f16(a, bm, acc);
            }
            const int o  = nt * 16 + (lane & 15);
            const float al = sAlpha[o], be = sBeta[o];
            const int rbase = mt * 16 + ((lane >> 4) << 3);
            for (int r = 0; r < 8; ++r) {
                float y = acc[r] * al + be;
                y = y > 0.0f ? y : 0.0f;
                sA1[(rbase + r) * WP + o] = (_Float16)y;
            }
        }
    }
    __syncthreads();

    // ---- layer 1: 32 -> 32 --------------------------------------------------
    for (int mi = 0; mi < 2; ++mi) {
        const int mt = 2 * w + mi;
        for (int nt = 0; nt < 2; ++nt) {
            v8f acc = {};
            const v16h a  = ldsA(sA1, WP, mt * 16, 0, lane);
            const v16h bm = ldsB(sW1, WP, nt * 16, 0, lane);
            acc = wmma_f16(a, bm, acc);
            const int o  = nt * 16 + (lane & 15);
            const float al = sAlpha[32 + o], be = sBeta[32 + o];
            const int rbase = mt * 16 + ((lane >> 4) << 3);
            for (int r = 0; r < 8; ++r) {
                float y = acc[r] * al + be;
                y = y > 0.0f ? y : 0.0f;
                sA2[(rbase + r) * WP + o] = (_Float16)y;
            }
        }
    }
    __syncthreads();

    // ---- layer 2: 32 -> 128, fused masked mean pool over the 32 samples ----
    const float qf  = (qmask[(size_t)b * N1C + n1] != 0) ? 1.0f : 0.0f;
    const int   cq  = cntIn[qg];
    const float den = (float)cq + (1.0f - qf) * 32.0f;  // 0 -> NaN (matches ref)
    const int   h   = lane >> 4;

    for (int nt = 0; nt < 8; ++nt) {
        const v16h bm = ldsB(sW2, WP, nt * 16, 0, lane);
        v8f acc0 = {}, acc1 = {};
        const v16h a0 = ldsA(sA2, WP, (2 * w) * 16, 0, lane);
        acc0 = wmma_f16(a0, bm, acc0);
        const v16h a1 = ldsA(sA2, WP, (2 * w + 1) * 16, 0, lane);
        acc1 = wmma_f16(a1, bm, acc1);

        const int o  = nt * 16 + (lane & 15);
        const float al = sAlpha[64 + o], be = sBeta[64 + o];
        float num = 0.0f;
        for (int r = 0; r < 8; ++r) {
            const int s0 = r + 8 * h;          // sample ids covered by acc0
            const int s1 = 16 + r + 8 * h;     // sample ids covered by acc1
            float y0 = acc0[r] * al + be; y0 = y0 > 0.0f ? y0 : 0.0f;
            float y1 = acc1[r] * al + be; y1 = y1 > 0.0f ? y1 : 0.0f;
            const float fm0 = ((s0 < cq) ? 1.0f : 0.0f) + (1.0f - qf);
            const float fm1 = ((s1 < cq) ? 1.0f : 0.0f) + (1.0f - qf);
            num += y0 * fm0 + y1 * fm1;
        }
        num += __shfl_xor(num, 16, 32);        // fold lane halves (M 0-7 vs 8-15)
        const float val = num / den;
        if (lane < 16)
            out[((size_t)b * COUT + o) * N1C + n1] = val;
    }
}

// ---------------------------------------------------------------------------
extern "C" void kernel_launch(void* const* d_in, const int* in_sizes, int n_in,
                              void* d_out, int out_size, void* d_ws, size_t ws_size,
                              hipStream_t stream)
{
    (void)in_sizes; (void)n_in; (void)out_size; (void)ws_size;
    const float* qxyz  = (const float*)d_in[0];
    const float* sxyz  = (const float*)d_in[1];
    const int*   qmask = (const int*)  d_in[2];
    const int*   smask = (const int*)  d_in[3];
    const float* feat  = (const float*)d_in[4];
    const float* W0  = (const float*)d_in[5];
    const float* g0  = (const float*)d_in[6];
    const float* b0  = (const float*)d_in[7];
    const float* rm0 = (const float*)d_in[8];
    const float* rv0 = (const float*)d_in[9];
    const float* W1  = (const float*)d_in[10];
    const float* g1  = (const float*)d_in[11];
    const float* b1  = (const float*)d_in[12];
    const float* rm1 = (const float*)d_in[13];
    const float* rv1 = (const float*)d_in[14];
    const float* W2  = (const float*)d_in[15];
    const float* g2  = (const float*)d_in[16];
    const float* b2  = (const float*)d_in[17];
    const float* rm2 = (const float*)d_in[18];
    const float* rv2 = (const float*)d_in[19];

    const int BQTOT = 2 * N1C;                 // B * N1 queries
    int* idx = (int*)d_ws;                     // [B*N1*32]
    int* cnt = idx + (size_t)BQTOT * NS;       // [B*N1]

    bq_kernel<<<BQTOT / 256, 256, 0, stream>>>(qxyz, sxyz, qmask, smask, idx, cnt);

    mlp_kernel<<<BQTOT / BQ, 128, 0, stream>>>(
        qxyz, sxyz, qmask, feat,
        W0, g0, b0, rm0, rv0,
        W1, g1, b1, rm1, rv1,
        W2, g2, b2, rm2, rv2,
        idx, cnt, (float*)d_out);
}
